// get_model_19061064860388
// MI455X (gfx1250) — compile-verified, matrix operand order checked by reading, not run on hardware
//
#include <hip/hip_runtime.h>
#include <hip/hip_bf16.h>
#include <math.h>

// ---------------------------------------------------------------------------
// CDNA5 (gfx1250) wave32 WMMA types/helpers
// ---------------------------------------------------------------------------
typedef __attribute__((ext_vector_type(16))) __bf16 v16bf;
typedef __attribute__((ext_vector_type(8)))  float  v8f;

__device__ __forceinline__ __bf16 f2bf(float f) {
  union { float f; unsigned u; } c; c.f = f;
  unsigned short h = (unsigned short)(c.u >> 16);
  __bf16 r; __builtin_memcpy(&r, &h, 2);
  return r;
}
__device__ __forceinline__ unsigned short f2bfbits(float f) {
  union { float f; unsigned u; } c; c.f = f;
  return (unsigned short)(c.u >> 16);
}

// 16-bit A/B fragment K-offset for lane-half kh (ISA 7.12.2):
// lanes 0-15 hold K = j (j<8) and 16+(j-8) (j>=8); lanes 16-31 are +8.
__device__ __forceinline__ int koff(int j, int kh) {
  return kh * 8 + (j < 8 ? j : j + 8);
}

// A fragment from f32 row: two groups of 8 consecutive floats (4 float4 loads).
__device__ __forceinline__ v16bf load_a_frag_f32(const float* __restrict__ row,
                                                 int kk, int kh) {
  const float4* p = (const float4*)(row + kk + kh * 8);
  float4 q0 = p[0], q1 = p[1], q2 = p[4], q3 = p[5];
  v16bf a;
  a[0]  = f2bf(q0.x); a[1]  = f2bf(q0.y); a[2]  = f2bf(q0.z); a[3]  = f2bf(q0.w);
  a[4]  = f2bf(q1.x); a[5]  = f2bf(q1.y); a[6]  = f2bf(q1.z); a[7]  = f2bf(q1.w);
  a[8]  = f2bf(q2.x); a[9]  = f2bf(q2.y); a[10] = f2bf(q2.z); a[11] = f2bf(q2.w);
  a[12] = f2bf(q3.x); a[13] = f2bf(q3.y); a[14] = f2bf(q3.z); a[15] = f2bf(q3.w);
  return a;
}

// A fragment from bf16 row: two aligned 16-byte loads, zero repack VALU.
__device__ __forceinline__ v16bf load_a_frag_bf(const unsigned short* __restrict__ row,
                                                int kk, int kh) {
  const uint4* pa = (const uint4*)(row + kk + kh * 8);
  uint4 lo = pa[0];          // K elements j=0..7
  uint4 hi = pa[2];          // K elements j=8..15 (+16 ushorts)
  struct { uint4 a, b; } tmp{lo, hi};
  v16bf a; __builtin_memcpy(&a, &tmp, 32);
  return a;
}

#define WAVES_PER_BLOCK 8

// ---------------------------------------------------------------------------
// Input split: x [4,6,2048] -> xyz0 [4,2048,3], pts0 [4,2048,3]
// ---------------------------------------------------------------------------
__global__ void split_kernel(const float* __restrict__ x,
                             float* __restrict__ xyz0, float* __restrict__ pts0) {
  int t = blockIdx.x * blockDim.x + threadIdx.x;
  const int total = 4 * 2048 * 3;
  if (t >= total) return;
  int d = t % 3;
  int n = (t / 3) % 2048;
  int b = t / (3 * 2048);
  xyz0[t] = x[((size_t)b * 6 + d) * 2048 + n];
  pts0[t] = x[((size_t)b * 6 + 3 + d) * 2048 + n];
}

__global__ void zero_kernel(float* __restrict__ p, long long n) {
  long long t = (long long)blockIdx.x * blockDim.x + threadIdx.x;
  if (t < n) p[t] = 0.0f;
}

__global__ void f32_to_bf16_kernel(const float* __restrict__ src,
                                   unsigned short* __restrict__ dst, long long n) {
  long long t = (long long)blockIdx.x * blockDim.x + threadIdx.x;
  if (t < n) dst[t] = f2bfbits(src[t]);
}

// ---------------------------------------------------------------------------
// Farthest point sampling: one block per batch (record far, then update dist)
// ---------------------------------------------------------------------------
__global__ void fps_kernel(const float* __restrict__ xyz, int Np, int S,
                           int* __restrict__ fidx, float* __restrict__ nxyz) {
  int b = blockIdx.x;
  __shared__ float dist[2048];
  __shared__ float rmax[256];
  __shared__ int   rarg[256];
  __shared__ int   sfar;
  const float* P = xyz + (size_t)b * Np * 3;
  for (int n = threadIdx.x; n < Np; n += blockDim.x) dist[n] = 1e10f;
  if (threadIdx.x == 0) sfar = 0;
  __syncthreads();
  for (int it = 0; it < S; ++it) {
    int far = sfar;
    float cx = P[far * 3 + 0], cy = P[far * 3 + 1], cz = P[far * 3 + 2];
    if (threadIdx.x == 0) {
      fidx[b * S + it] = far;
      nxyz[((size_t)b * S + it) * 3 + 0] = cx;
      nxyz[((size_t)b * S + it) * 3 + 1] = cy;
      nxyz[((size_t)b * S + it) * 3 + 2] = cz;
    }
    float best = -1.0f; int bi = 0;
    for (int n = threadIdx.x; n < Np; n += blockDim.x) {
      float dx = P[n * 3 + 0] - cx;
      float dy = P[n * 3 + 1] - cy;
      float dz = P[n * 3 + 2] - cz;
      float d = dx * dx + dy * dy + dz * dz;
      float dm = fminf(dist[n], d);
      dist[n] = dm;
      if (dm > best) { best = dm; bi = n; }
    }
    rmax[threadIdx.x] = best; rarg[threadIdx.x] = bi;
    __syncthreads();
    for (int st = blockDim.x >> 1; st; st >>= 1) {
      if (threadIdx.x < st) {
        float o = rmax[threadIdx.x + st]; int oi = rarg[threadIdx.x + st];
        if (o > rmax[threadIdx.x] || (o == rmax[threadIdx.x] && oi < rarg[threadIdx.x])) {
          rmax[threadIdx.x] = o; rarg[threadIdx.x] = oi;
        }
      }
      __syncthreads();
    }
    if (threadIdx.x == 0) sfar = rarg[0];
    __syncthreads();
  }
}

// ---------------------------------------------------------------------------
// Ball query: first K ascending indices within radius; pad with first hit.
// ---------------------------------------------------------------------------
__global__ void ball_query_kernel(const float* __restrict__ xyz,
                                  const float* __restrict__ nxyz,
                                  int Np, int S, int Kq, float r2,
                                  int* __restrict__ gidx) {
  int t = blockIdx.x * blockDim.x + threadIdx.x;
  if (t >= 4 * S) return;
  int b = t / S;
  const float* P = xyz + (size_t)b * Np * 3;
  float cx = nxyz[(size_t)t * 3 + 0];
  float cy = nxyz[(size_t)t * 3 + 1];
  float cz = nxyz[(size_t)t * 3 + 2];
  int* out = gidx + (size_t)t * Kq;
  int cnt = 0, first = 0;
  for (int n = 0; n < Np && cnt < Kq; ++n) {
    float dx = P[n * 3 + 0] - cx;
    float dy = P[n * 3 + 1] - cy;
    float dz = P[n * 3 + 2] - cz;
    float d2 = dx * dx + dy * dy + dz * dz;
    if (d2 <= r2) { if (cnt == 0) first = n; out[cnt++] = n; }
  }
  for (; cnt < Kq; ++cnt) out[cnt] = first;
}

// ---------------------------------------------------------------------------
// BN folding directly into WMMA B-fragment order:
//   out[((kt*ntn + nt)*32 + lane)*16 + j] = bf16( W[co][k] * gamma/sqrt(var+eps) )
// ---------------------------------------------------------------------------
__global__ void fold_w_kernel(const float* __restrict__ W, const float* __restrict__ g,
                              const float* __restrict__ vr, int cin, int cout,
                              int Kp, int ntn, unsigned short* __restrict__ wf) {
  long long t = (long long)blockIdx.x * blockDim.x + threadIdx.x;
  long long total = (long long)Kp * ntn * 16;
  if (t >= total) return;
  int j  = (int)(t & 15);
  int l  = (int)((t >> 4) & 31);
  long long rest = t >> 9;
  int nt = (int)(rest % ntn);
  int kt = (int)(rest / ntn);
  int k  = kt * 32 + koff(j, l >> 4);
  int co = nt * 16 + (l & 15);
  float v = 0.0f;
  if (k < cin && co < cout) {
    float sc = g[co] * rsqrtf(vr[co] + 1e-5f);
    v = W[(size_t)co * cin + k] * sc;
  }
  wf[t] = f2bfbits(v);
}

__global__ void fold_b_kernel(const float* __restrict__ bb, const float* __restrict__ g,
                              const float* __restrict__ be, const float* __restrict__ mn,
                              const float* __restrict__ vr, int cout, int Ns,
                              float* __restrict__ bf) {
  int i = blockIdx.x * blockDim.x + threadIdx.x;
  if (i >= Ns) return;
  float v = 0.0f;
  if (i < cout) {
    float sc = g[i] * rsqrtf(vr[i] + 1e-5f);
    v = (bb[i] - mn[i]) * sc + be[i];
  }
  bf[i] = v;
}

// Generic f32 [K x N] -> bf16 fragment swizzle (for activation B-matrices).
__global__ void swizzle_b_kernel(const float* __restrict__ Bsrc, int ldb, int ntn,
                                 long long total, unsigned short* __restrict__ outw) {
  long long t = (long long)blockIdx.x * blockDim.x + threadIdx.x;
  if (t >= total) return;
  int j  = (int)(t & 15);
  int l  = (int)((t >> 4) & 31);
  long long rest = t >> 9;
  int nt = (int)(rest % ntn);
  int kt = (int)(rest / ntn);
  int k  = kt * 32 + koff(j, l >> 4);
  int col = nt * 16 + (l & 15);
  outw[t] = f2bfbits(Bsrc[(size_t)k * ldb + col]);
}

// ---------------------------------------------------------------------------
// WMMA GEMM with pairwise two-set register double buffering (no rotation
// copies, no WAR on in-flight WMMA operands). A f32/bf16; out f32/bf16.
// ---------------------------------------------------------------------------
template <int TN, int ABF, int OBF>
__global__ __launch_bounds__(256) void
wmma_gemm_t(const void* __restrict__ Ap, int lda,
            const unsigned short* __restrict__ wf,
            const float* __restrict__ bias,
            void* __restrict__ Yp, int ldy,
            int M, int Kd, int ntn, int relu) {
  const int lane = threadIdx.x & 31;
  const int wid  = blockIdx.x * WAVES_PER_BLOCK + (threadIdx.x >> 5);
  const int ng = ntn / TN, mt = M >> 4;
  if (wid >= mt * ng) return;
  const int tm  = wid / ng;
  const int tnb = (wid % ng) * TN;
  const int r16 = lane & 15, kh = lane >> 4;
  const float*          Afr = (const float*)Ap + (size_t)(tm * 16 + r16) * lda;
  const unsigned short* Abr = (const unsigned short*)Ap + (size_t)(tm * 16 + r16) * lda;
  const v16bf* Bf = (const v16bf*)wf;
  v8f acc[TN];
#pragma unroll
  for (int t = 0; t < TN; ++t) acc[t] = v8f{};

#define LOAD_SET(KKK, AV, BV)                                               \
  do {                                                                      \
    if constexpr (ABF) AV = load_a_frag_bf(Abr, (KKK), kh);                 \
    else               AV = load_a_frag_f32(Afr, (KKK), kh);                \
    size_t fb_ = ((size_t)((KKK) >> 5) * ntn + tnb) * 32 + lane;            \
    _Pragma("unroll")                                                       \
    for (int t_ = 0; t_ < TN; ++t_) BV[t_] = Bf[fb_ + (size_t)t_ * 32];     \
  } while (0)

  const int nk = Kd >> 5;
  v16bf a0, a1, b0[TN], b1[TN];
  LOAD_SET(0, a0, b0);
  int i = 0;
  for (; i + 2 <= nk; i += 2) {
    LOAD_SET((i + 1) << 5, a1, b1);
#pragma unroll
    for (int t = 0; t < TN; ++t)
      acc[t] = __builtin_amdgcn_wmma_f32_16x16x32_bf16(false, a0, false, b0[t],
                                                       (short)0, acc[t], false, false);
    if (i + 2 < nk) LOAD_SET((i + 2) << 5, a0, b0);
#pragma unroll
    for (int t = 0; t < TN; ++t)
      acc[t] = __builtin_amdgcn_wmma_f32_16x16x32_bf16(false, a1, false, b1[t],
                                                       (short)0, acc[t], false, false);
  }
  if (i < nk) {
#pragma unroll
    for (int t = 0; t < TN; ++t)
      acc[t] = __builtin_amdgcn_wmma_f32_16x16x32_bf16(false, a0, false, b0[t],
                                                       (short)0, acc[t], false, false);
  }
#undef LOAD_SET

#pragma unroll
  for (int t = 0; t < TN; ++t) {
    int col = (tnb + t) * 16 + r16;
    float bv = bias[col];
#pragma unroll
    for (int r = 0; r < 8; ++r) {
      int row = tm * 16 + r + kh * 8;
      float y = acc[t][r] + bv;
      if (relu) y = fmaxf(y, 0.0f);
      if constexpr (OBF) ((unsigned short*)Yp)[(size_t)row * ldy + col] = f2bfbits(y);
      else               ((float*)Yp)[(size_t)row * ldy + col] = y;
    }
  }
}

// conv1 of each branch: A rows gathered on the fly (pts ++ relative xyz ++ 0),
// bf16 output.
template <int TN>
__global__ __launch_bounds__(256) void
wmma_gemm_gather_t(const float* __restrict__ pts, int cf, int Np,
                   const float* __restrict__ xyz,
                   const float* __restrict__ nxyz,
                   const int* __restrict__ gidx,
                   int S, int Kg,
                   const unsigned short* __restrict__ wf,
                   const float* __restrict__ bias,
                   unsigned short* __restrict__ Y, int ldy,
                   int M, int Kp, int ntn) {
  const int lane = threadIdx.x & 31;
  const int wid  = blockIdx.x * WAVES_PER_BLOCK + (threadIdx.x >> 5);
  const int ng = ntn / TN, mt = M >> 4;
  if (wid >= mt * ng) return;
  const int tm  = wid / ng;
  const int tnb = (wid % ng) * TN;
  const int r16 = lane & 15, kh = lane >> 4;
  const int arow = tm * 16 + r16;
  int g   = arow / Kg;
  int kpt = arow - g * Kg;
  int b   = g / S;
  int idx = gidx[(size_t)g * Kg + kpt];
  const float* prow = pts  + ((size_t)b * Np + idx) * cf;
  const float* xrow = xyz  + ((size_t)b * Np + idx) * 3;
  const float* crow = nxyz + (size_t)g * 3;
  const v16bf* Bf = (const v16bf*)wf;
  v8f acc[TN];
#pragma unroll
  for (int t = 0; t < TN; ++t) acc[t] = v8f{};
  for (int kk = 0; kk < Kp; kk += 32) {
    v16bf a;
#pragma unroll
    for (int j = 0; j < 16; ++j) {
      int k = kk + koff(j, kh);
      float v;
      if (k < cf)          v = prow[k];
      else if (k < cf + 3) v = xrow[k - cf] - crow[k - cf];
      else                 v = 0.0f;
      a[j] = f2bf(v);
    }
    size_t fb = ((size_t)(kk >> 5) * ntn + tnb) * 32 + lane;
#pragma unroll
    for (int t = 0; t < TN; ++t) {
      v16bf bb = Bf[fb + (size_t)t * 32];
      acc[t] = __builtin_amdgcn_wmma_f32_16x16x32_bf16(false, a, false, bb,
                                                       (short)0, acc[t], false, false);
    }
  }
#pragma unroll
  for (int t = 0; t < TN; ++t) {
    int col = (tnb + t) * 16 + r16;
    float bv = bias[col];
#pragma unroll
    for (int r = 0; r < 8; ++r) {
      int row = tm * 16 + r + kh * 8;
      Y[(size_t)row * ldy + col] = f2bfbits(fmaxf(acc[t][r] + bv, 0.0f));
    }
  }
}

// last conv of each branch: bf16 A, relu + fused max-pool over group dimension
// (post-ReLU values >= 0, so unsigned-bits atomicMax is order-correct).
template <int TN>
__global__ __launch_bounds__(256) void
wmma_gemm_pool_t(const unsigned short* __restrict__ A, int lda,
                 const unsigned short* __restrict__ wf,
                 const float* __restrict__ bias,
                 int M, int Kd, int ntn, int Kg,
                 float* __restrict__ outp, int Ctot, int coff) {
  const int lane = threadIdx.x & 31;
  const int wid  = blockIdx.x * WAVES_PER_BLOCK + (threadIdx.x >> 5);
  const int ng = ntn / TN, mt = M >> 4;
  if (wid >= mt * ng) return;
  const int tm  = wid / ng;
  const int tnb = (wid % ng) * TN;
  const int r16 = lane & 15, kh = lane >> 4;
  const unsigned short* Abr = A + (size_t)(tm * 16 + r16) * lda;
  const v16bf* Bf = (const v16bf*)wf;
  v8f acc[TN];
#pragma unroll
  for (int t = 0; t < TN; ++t) acc[t] = v8f{};

#define LOAD_SET(KKK, AV, BV)                                               \
  do {                                                                      \
    AV = load_a_frag_bf(Abr, (KKK), kh);                                    \
    size_t fb_ = ((size_t)((KKK) >> 5) * ntn + tnb) * 32 + lane;            \
    _Pragma("unroll")                                                       \
    for (int t_ = 0; t_ < TN; ++t_) BV[t_] = Bf[fb_ + (size_t)t_ * 32];     \
  } while (0)

  const int nk = Kd >> 5;
  v16bf a0, a1, b0[TN], b1[TN];
  LOAD_SET(0, a0, b0);
  int i = 0;
  for (; i + 2 <= nk; i += 2) {
    LOAD_SET((i + 1) << 5, a1, b1);
#pragma unroll
    for (int t = 0; t < TN; ++t)
      acc[t] = __builtin_amdgcn_wmma_f32_16x16x32_bf16(false, a0, false, b0[t],
                                                       (short)0, acc[t], false, false);
    if (i + 2 < nk) LOAD_SET((i + 2) << 5, a0, b0);
#pragma unroll
    for (int t = 0; t < TN; ++t)
      acc[t] = __builtin_amdgcn_wmma_f32_16x16x32_bf16(false, a1, false, b1[t],
                                                       (short)0, acc[t], false, false);
  }
  if (i < nk) {
#pragma unroll
    for (int t = 0; t < TN; ++t)
      acc[t] = __builtin_amdgcn_wmma_f32_16x16x32_bf16(false, a0, false, b0[t],
                                                       (short)0, acc[t], false, false);
  }
#undef LOAD_SET

#pragma unroll
  for (int t = 0; t < TN; ++t) {
    int col = (tnb + t) * 16 + r16;
    float bv = bias[col];
#pragma unroll
    for (int r = 0; r < 8; ++r) {
      int row = tm * 16 + r + kh * 8;
      float y = fmaxf(acc[t][r] + bv, 0.0f);
      int g = row / Kg;
      unsigned* dst = (unsigned*)&outp[(size_t)g * Ctot + coff + col];
      atomicMax(dst, __float_as_uint(y));
    }
  }
}

// feat = weights[b] (112x512 bf16 rows, rows>=100 ignored) x ctx[b] (preswizzled)
__global__ __launch_bounds__(256) void
wmma_gemm_feat(const unsigned short* __restrict__ Wgbf,
               const unsigned short* __restrict__ ctxw,
               float* __restrict__ feat) {
  const int lane = threadIdx.x & 31;
  const int wid  = blockIdx.x * WAVES_PER_BLOCK + (threadIdx.x >> 5);
  const int ng = 16, mt = 28;           // M=448 (4*112), N=512 (TN=2), K=512
  if (wid >= mt * ng) return;
  const int tm  = wid / ng;
  const int tnb = (wid % ng) * 2;
  const int r16 = lane & 15, kh = lane >> 4;
  const int b = tm / 7;                 // 7 M-tiles per batch (112/16)
  const unsigned short* Abr = Wgbf + (size_t)(tm * 16 + r16) * 512;
  const v16bf* Bf = (const v16bf*)(ctxw + (size_t)b * 262144);
  v8f acc[2];
#pragma unroll
  for (int t = 0; t < 2; ++t) acc[t] = v8f{};

#define LOAD_SET(KKK, AV, BV)                                               \
  do {                                                                      \
    AV = load_a_frag_bf(Abr, (KKK), kh);                                    \
    size_t fb_ = ((size_t)((KKK) >> 5) * 32 + tnb) * 32 + lane;             \
    _Pragma("unroll")                                                       \
    for (int t_ = 0; t_ < 2; ++t_) BV[t_] = Bf[fb_ + (size_t)t_ * 32];      \
  } while (0)

  const int nk = 16;                    // 512 / 32
  v16bf a0, a1, b0[2], b1[2];
  LOAD_SET(0, a0, b0);
  for (int i = 0; i + 2 <= nk; i += 2) {
    LOAD_SET((i + 1) << 5, a1, b1);
#pragma unroll
    for (int t = 0; t < 2; ++t)
      acc[t] = __builtin_amdgcn_wmma_f32_16x16x32_bf16(false, a0, false, b0[t],
                                                       (short)0, acc[t], false, false);
    if (i + 2 < nk) LOAD_SET((i + 2) << 5, a0, b0);
#pragma unroll
    for (int t = 0; t < 2; ++t)
      acc[t] = __builtin_amdgcn_wmma_f32_16x16x32_bf16(false, a1, false, b1[t],
                                                       (short)0, acc[t], false, false);
  }
#undef LOAD_SET

#pragma unroll
  for (int t = 0; t < 2; ++t) {
    int col = (tnb + t) * 16 + r16;
#pragma unroll
    for (int r = 0; r < 8; ++r) {
      int row = tm * 16 + r + kh * 8;
      int bb = row / 112, kidx = row % 112;
      if (kidx < 100)
        feat[((size_t)bb * 100 + kidx) * 512 + col] = acc[t][r];
    }
  }
}

// ---------------------------------------------------------------------------
// Softmax over the 512-point axis for each (b, k<100); weights [B,112,512].
// ---------------------------------------------------------------------------
__global__ void softmax_kernel(const float* __restrict__ h, int ldh,
                               float* __restrict__ wgt) {
  int b = blockIdx.x / 100, k = blockIdx.x % 100;
  const float* H = h + (size_t)b * 512 * ldh + k;
  __shared__ float red[256];
  int tid = threadIdx.x;
  float m = -3.4e38f;
  for (int s = tid; s < 512; s += 256) m = fmaxf(m, H[(size_t)s * ldh]);
  red[tid] = m; __syncthreads();
  for (int st = 128; st; st >>= 1) {
    if (tid < st) red[tid] = fmaxf(red[tid], red[tid + st]);
    __syncthreads();
  }
  m = red[0]; __syncthreads();
  float sum = 0.0f;
  for (int s = tid; s < 512; s += 256) sum += __expf(H[(size_t)s * ldh] - m);
  red[tid] = sum; __syncthreads();
  for (int st = 128; st; st >>= 1) {
    if (tid < st) red[tid] += red[tid + st];
    __syncthreads();
  }
  float inv = 1.0f / red[0];
  for (int s = tid; s < 512; s += 256)
    wgt[((size_t)b * 112 + k) * 512 + s] = __expf(H[(size_t)s * ldh] - m) * inv;
}

__global__ void skel_xyz_kernel(const float* __restrict__ wgt,
                                const float* __restrict__ xyz4,
                                float* __restrict__ skel) {
  int t = blockIdx.x * blockDim.x + threadIdx.x;   // [b,k,d], 1200 total
  if (t >= 4 * 100 * 3) return;
  int d = t % 3, k = (t / 3) % 100, b = t / 300;
  const float* w = wgt + ((size_t)b * 112 + k) * 512;
  const float* X = xyz4 + (size_t)b * 512 * 3 + d;
  float acc = 0.0f;
  for (int s = 0; s < 512; ++s) acc += w[s] * X[(size_t)s * 3];
  skel[t] = acc;
}

__global__ void min_d_kernel(const float* __restrict__ xyz4,
                             const float* __restrict__ skel,
                             float* __restrict__ mind) {
  int t = blockIdx.x * blockDim.x + threadIdx.x;   // [b,s], 2048 total
  if (t >= 4 * 512) return;
  int b = t / 512;
  float px = xyz4[(size_t)t * 3 + 0];
  float py = xyz4[(size_t)t * 3 + 1];
  float pz = xyz4[(size_t)t * 3 + 2];
  float best = 3.4e38f;
  const float* S = skel + (size_t)b * 100 * 3;
  for (int k = 0; k < 100; ++k) {
    float dx = px - S[k * 3 + 0];
    float dy = py - S[k * 3 + 1];
    float dz = pz - S[k * 3 + 2];
    best = fminf(best, sqrtf(dx * dx + dy * dy + dz * dz));
  }
  mind[t] = best;
}

__global__ void skel_r_kernel(const float* __restrict__ wgt,
                              const float* __restrict__ mind,
                              float* __restrict__ skel_r) {
  int t = blockIdx.x * blockDim.x + threadIdx.x;   // [b,k], 400 total
  if (t >= 4 * 100) return;
  int b = t / 100, k = t % 100;
  const float* w = wgt + ((size_t)b * 112 + k) * 512;
  const float* m = mind + (size_t)b * 512;
  float acc = 0.0f;
  for (int s = 0; s < 512; ++s) acc += w[s] * m[s];
  skel_r[t] = acc;
}

// ---------------------------------------------------------------------------
// Host orchestration
// ---------------------------------------------------------------------------
struct ConvP { const float *W, *b, *g, *be, *mn, *vr; };

static inline int cdiv(long long a, long long b) { return (int)((a + b - 1) / b); }
static inline int al32(int x) { return (x + 31) & ~31; }

extern "C" void kernel_launch(void* const* d_in, const int* in_sizes, int n_in,
                              void* d_out, int out_size, void* d_ws, size_t ws_size,
                              hipStream_t stream) {
  (void)in_sizes; (void)n_in; (void)out_size; (void)ws_size;
  const float* xin = (const float*)d_in[0];
  int p = 1;
  auto nextConv = [&](ConvP& c) {
    c.W  = (const float*)d_in[p++]; c.b  = (const float*)d_in[p++];
    c.g  = (const float*)d_in[p++]; c.be = (const float*)d_in[p++];
    c.mn = (const float*)d_in[p++]; c.vr = (const float*)d_in[p++];
  };
  ConvP sa[4][2][3], hd[5];
  for (int l = 0; l < 4; ++l)
    for (int br = 0; br < 2; ++br)
      for (int c = 0; c < 3; ++c) nextConv(sa[l][br][c]);
  for (int i = 0; i < 5; ++i) nextConv(hd[i]);

  // ---- workspace layout (floats, 32B-aligned blocks) ----
  float* wsf = (float*)d_ws;
  size_t off = 0;
  auto alloc = [&](size_t n) {
    float* r = wsf + off;
    off += (n + 7) & ~(size_t)7;
    return r;
  };

  const int Nin[5]    = {2048, 1024, 768, 512, 512};
  const int Sout[4]   = {1024, 768, 512, 512};
  const int inch[4]   = {3, 64, 128, 256};
  const int KK[4][2]  = {{16, 32}, {32, 64}, {32, 64}, {64, 128}};
  const float RR[4][2] = {{0.1f, 0.2f}, {0.2f, 0.4f}, {0.4f, 0.6f}, {0.6f, 0.8f}};
  const int MLP[4][3] = {{16, 16, 32}, {32, 32, 64}, {64, 64, 128}, {128, 128, 256}};
  const int CTOT[4]   = {64, 128, 256, 512};

  float* xyzL[5]; float* ptsL[5];
  xyzL[0] = alloc((size_t)4 * 2048 * 3);
  ptsL[0] = alloc((size_t)4 * 2048 * 3);
  for (int l = 0; l < 4; ++l) {
    xyzL[l + 1] = alloc((size_t)4 * Sout[l] * 3);
    ptsL[l + 1] = alloc((size_t)4 * Sout[l] * CTOT[l]);
  }
  int*            fidx  = (int*)alloc(4 * 1024);
  int*            gidx  = (int*)alloc((size_t)4 * 512 * 128);
  unsigned short* wfold = (unsigned short*)alloc((size_t)512 * 512 / 2);
  float*          bfold = alloc(512);
  unsigned short* ping  = (unsigned short*)alloc((size_t)262144 * 128 / 2);  // bf16 acts
  unsigned short* pong  = (unsigned short*)alloc((size_t)262144 * 128 / 2);
  float*          wbuf  = alloc((size_t)4 * 112 * 512);
  unsigned short* wbufb = (unsigned short*)alloc((size_t)4 * 112 * 512 / 2);
  float*          mind  = alloc((size_t)4 * 512);
  float*          hfin  = alloc((size_t)2048 * 112);
  unsigned short* ctxw  = (unsigned short*)alloc((size_t)4 * 262144 / 2);

  auto fold = [&](const ConvP& c, int cin, int cout, int Kp, int Ns) {
    fold_w_kernel<<<cdiv((long long)Kp * Ns, 256), 256, 0, stream>>>(
        c.W, c.g, c.vr, cin, cout, Kp, Ns >> 4, wfold);
    fold_b_kernel<<<cdiv(Ns, 256), 256, 0, stream>>>(
        c.b, c.g, c.be, c.mn, c.vr, cout, Ns, bfold);
  };
  auto pickTN = [](int Ns) { return (Ns % 64 == 0) ? 4 : ((Ns % 32 == 0) ? 2 : 1); };
  auto blocks_for = [&](int M, int Ns, int TN) {
    return cdiv((long long)(M >> 4) * ((Ns >> 4) / TN), WAVES_PER_BLOCK);
  };

  // plain GEMM dispatch over (TN, A-bf16, OUT-bf16); f32-A capped at TN=2
  auto launch_plain = [&](const void* A, int lda, void* Y, int ldy,
                          int M, int Kd, int Ns, int relu, int abf, int obf) {
    int TN = pickTN(Ns);
    if (!abf && TN == 4) TN = 2;            // f32-A staging: keep VGPR pressure down
    int nb = blocks_for(M, Ns, TN), ntn = Ns >> 4;
#define GEMM_CASE(TN_, AB_, OB_)                                                 \
    wmma_gemm_t<TN_, AB_, OB_><<<nb, 256, 0, stream>>>(A, lda, wfold, bfold, Y,  \
                                                       ldy, M, Kd, ntn, relu)
    if (TN == 4) {
      if (abf) { if (obf) GEMM_CASE(4, 1, 1); else GEMM_CASE(4, 1, 0); }
      else     { if (obf) GEMM_CASE(4, 0, 1); else GEMM_CASE(4, 0, 0); }
    } else if (TN == 2) {
      if (abf) { if (obf) GEMM_CASE(2, 1, 1); else GEMM_CASE(2, 1, 0); }
      else     { if (obf) GEMM_CASE(2, 0, 1); else GEMM_CASE(2, 0, 0); }
    } else {
      if (abf) { if (obf) GEMM_CASE(1, 1, 1); else GEMM_CASE(1, 1, 0); }
      else     { if (obf) GEMM_CASE(1, 0, 1); else GEMM_CASE(1, 0, 0); }
    }
#undef GEMM_CASE
  };

  // ---- input split ----
  split_kernel<<<cdiv(4 * 2048 * 3, 256), 256, 0, stream>>>(xin, xyzL[0], ptsL[0]);

  // ---- set-abstraction layers ----
  for (int l = 0; l < 4; ++l) {
    const int Np = Nin[l], S = Sout[l];
    fps_kernel<<<4, 256, 0, stream>>>(xyzL[l], Np, S, fidx, xyzL[l + 1]);
    zero_kernel<<<cdiv((long long)4 * S * CTOT[l], 256), 256, 0, stream>>>(
        ptsL[l + 1], (long long)4 * S * CTOT[l]);
    int coff = 0;
    for (int br = 0; br < 2; ++br) {
      const int Kg = KK[l][br];
      const float rad = RR[l][br];
      ball_query_kernel<<<cdiv(4 * S, 256), 256, 0, stream>>>(
          xyzL[l], xyzL[l + 1], Np, S, Kg, rad * rad, gidx);
      const int M = 4 * S * Kg;
      const int cinT = inch[l] + 3, Kp = al32(cinT);
      const int C1 = MLP[l][0], Ns1 = al32(C1);
      const int C2 = MLP[l][1], Ns2 = al32(C2);
      const int C3 = MLP[l][2], Ns3 = al32(C3);
      // conv1: gathered A -> bf16 activations
      fold(sa[l][br][0], cinT, C1, Kp, Ns1);
      {
        int TN = pickTN(Ns1), nb = blocks_for(M, Ns1, TN), ntn = Ns1 >> 4;
        if (TN == 4)
          wmma_gemm_gather_t<4><<<nb, 256, 0, stream>>>(
              ptsL[l], inch[l], Np, xyzL[l], xyzL[l + 1], gidx, S, Kg,
              wfold, bfold, ping, Ns1, M, Kp, ntn);
        else
          wmma_gemm_gather_t<2><<<nb, 256, 0, stream>>>(
              ptsL[l], inch[l], Np, xyzL[l], xyzL[l + 1], gidx, S, Kg,
              wfold, bfold, ping, Ns1, M, Kp, ntn);
      }
      // conv2: bf16 -> bf16
      fold(sa[l][br][1], C1, C2, Ns1, Ns2);
      launch_plain(ping, Ns1, pong, Ns2, M, Ns1, Ns2, 1, 1, 1);
      // conv3: bf16 A, fused relu + max-pool (f32 atomic out)
      fold(sa[l][br][2], C2, C3, Ns2, Ns3);
      {
        int TN = pickTN(Ns3), nb = blocks_for(M, Ns3, TN), ntn = Ns3 >> 4;
        if (TN == 4)
          wmma_gemm_pool_t<4><<<nb, 256, 0, stream>>>(
              pong, Ns2, wfold, bfold, M, Ns2, ntn, Kg, ptsL[l + 1], CTOT[l], coff);
        else
          wmma_gemm_pool_t<2><<<nb, 256, 0, stream>>>(
              pong, Ns2, wfold, bfold, M, Ns2, ntn, Kg, ptsL[l + 1], CTOT[l], coff);
      }
      coff += C3;
    }
  }

  // ---- head MLP on ctx = ptsL[4] [B*512, 512] ----
  const int hc[6] = {512, 384, 256, 256, 128, 100};
  const void* hin = ptsL[4];
  int lin = 512, abf = 0;
  void* hb[2] = {ping, pong};
  for (int i = 0; i < 4; ++i) {
    const int ci = hc[i], co = hc[i + 1];
    const int Ns = al32(co);
    fold(hd[i], ci, co, ci, Ns);
    launch_plain(hin, lin, hb[i & 1], Ns, 2048, ci, Ns, 1, abf, 1);
    hin = hb[i & 1]; lin = Ns; abf = 1;
  }
  fold(hd[4], 128, 100, 128, 112);
  launch_plain(hin, 128, hfin, 112, 2048, 128, 112, 0, 1, 0);

  // ---- softmax weights + skeleton outputs ----
  zero_kernel<<<cdiv((long long)4 * 112 * 512, 256), 256, 0, stream>>>(
      wbuf, (long long)4 * 112 * 512);
  softmax_kernel<<<400, 256, 0, stream>>>(hfin, 112, wbuf);

  float* out = (float*)d_out;          // skel_xyz[1200] | skel_r[400] | feat[204800]
  skel_xyz_kernel<<<cdiv(1200, 256), 256, 0, stream>>>(wbuf, xyzL[4], out);
  min_d_kernel<<<cdiv(2048, 256), 256, 0, stream>>>(xyzL[4], out, mind);
  skel_r_kernel<<<cdiv(400, 256), 256, 0, stream>>>(wbuf, mind, out + 1200);

  // feat GEMM: bf16 weights rows + preswizzled bf16 ctx fragments
  f32_to_bf16_kernel<<<cdiv((long long)4 * 112 * 512, 256), 256, 0, stream>>>(
      wbuf, wbufb, (long long)4 * 112 * 512);
  for (int b = 0; b < 4; ++b)
    swizzle_b_kernel<<<cdiv((long long)262144, 256), 256, 0, stream>>>(
        ptsL[4] + (size_t)b * 512 * 512, 512, 32, 262144, ctxw + (size_t)b * 262144);
  wmma_gemm_feat<<<cdiv((long long)28 * 16, WAVES_PER_BLOCK), 256, 0, stream>>>(
      wbufb, ctxw, out + 1600);
}
